// MetaR_10350871183900
// MI455X (gfx1250) — compile-verified, matrix operand order checked by reading, not run on hardware
//
#include <hip/hip_runtime.h>
#include <hip/hip_bf16.h>

#define BB 512
#define FEW 5
#define NBR 200
#define DD 100
#define HHID 450
#define G4 1800   // 4*H
#define SEQ 5
#define NTOT 50

typedef __attribute__((ext_vector_type(16))) _Float16 v16h;
typedef __attribute__((ext_vector_type(8)))  _Float16 v8h;
typedef __attribute__((ext_vector_type(8)))  float    v8f;
typedef __attribute__((ext_vector_type(4)))  unsigned int v4u;
typedef __attribute__((ext_vector_type(8)))  int      v8i_;
typedef __attribute__((ext_vector_type(4)))  int      v4i_;

// Padded K sizes (multiples of 32)
#define KP_GCN   224   // 200
#define KP_IH0   224   // 200
#define KP_HH    480   // 450
#define KP_IH1   928   // 900
#define KP_OUT   1824  // 1800
#define NP_W     1808  // 1800 rows padded to 16
#define NP_SMALL 112   // 100 rows padded to 16
#define LDA_SUP  (FEW * KP_IH0)   // 1120
#define GCNW_ELEMS (NP_SMALL * KP_GCN)   // 25088 f16 elements (50176 B)

// Build a WMMA 16-bit operand fragment from padded f16 memory.
// Per ISA 7.12.2: lane(16*half+i) VGPR j<4 -> K = k0+2j+8*half (pairs),
// j>=4 -> K = k0+16+2(j-4)+8*half. Both chunks are 8 contiguous f16 (16B).
__device__ __forceinline__ v16h frag_f16(const _Float16* p) {
  v8h c0 = *(const v8h*)p;
  v8h c1 = *(const v8h*)(p + 16);
  v16h r;
#pragma unroll
  for (int i = 0; i < 8; ++i) { r[i] = c0[i]; r[i + 8] = c1[i]; }
  return r;
}

__device__ __forceinline__ float wave_red_sum(float x) {
#pragma unroll
  for (int o = 16; o > 0; o >>= 1) x += __shfl_xor(x, o, 32);
  return x;
}

// ---------------------------------------------------------------------------
// Convert/zero-pad an f32 (rows x K) matrix into f16 (rowsP x ldk).
// ---------------------------------------------------------------------------
__global__ void cvt_pad_f16(const float* __restrict__ src, int rows, int K,
                            _Float16* __restrict__ dst, int rowsP, int ldk) {
  int idx = blockIdx.x * blockDim.x + threadIdx.x;
  if (idx >= rowsP * ldk) return;
  int r = idx / ldk, k = idx % ldk;
  float v = (r < rows && k < K) ? src[(long)r * K + k] : 0.0f;
  dst[idx] = (_Float16)v;
}

// ---------------------------------------------------------------------------
// WMMA GEMM on padded f16 operands:
//   C[m][n] = sum_k A16[m*lda+k]*W16[n*ldw+k] + bias1[n]+bias2[n]+Cadd[m*ldadd+n]
// Requirements: M % 16 == 0, Kp % 32 == 0, W16 has >= ceil16(N) zero-padded rows.
// One 16x16 tile per wave; inner loop = 4x b128 loads + 1 wmma.
// ---------------------------------------------------------------------------
__global__ void gemm_f16wmma(const _Float16* __restrict__ A16, int lda,
                             const _Float16* __restrict__ W16, int ldw,
                             const float* __restrict__ bias1,
                             const float* __restrict__ bias2,
                             const float* Cadd, int ldadd,
                             float* C, int ldc,
                             int M, int N, int Kp) {
  int lane = threadIdx.x & 31;
  int wave = threadIdx.x >> 5;
  int tile = blockIdx.x * 8 + wave;
  int ntN = (N + 15) >> 4;
  int ntM = M >> 4;
  if (tile >= ntM * ntN) return;
  int mt = tile / ntN, nt = tile % ntN;
  int half = lane >> 4, l16 = lane & 15;
  const _Float16* pa = A16 + (long)(mt * 16 + l16) * lda + 8 * half;
  const _Float16* pw = W16 + (long)(nt * 16 + l16) * ldw + 8 * half;
  v8f acc = {};
  for (int ks = Kp >> 5; ks > 0; --ks) {
    __builtin_prefetch(pa + 128, 0, 1);
    v16h a = frag_f16(pa);
    v16h b = frag_f16(pw);
    acc = __builtin_amdgcn_wmma_f32_16x16x32_f16(false, a, false, b,
                                                 (short)0, acc, false, false);
    pa += 32;
    pw += 32;
  }
  int ncol = nt * 16 + l16;
  if (ncol < N) {
    float bn = (bias1 ? bias1[ncol] : 0.0f) + (bias2 ? bias2[ncol] : 0.0f);
#pragma unroll
    for (int r = 0; r < 8; ++r) {
      int mrow = mt * 16 + r + 8 * half;
      float v = acc[r] + bn;
      if (Cadd) v += Cadd[(long)mrow * ldadd + ncol];
      C[(long)mrow * ldc + ncol] = v;
    }
  }
}

// ---------------------------------------------------------------------------
// Neighbor encoder: one block per (side, f, b) instance.
// gcn_W (padded f16, 50KB) is staged into LDS once per block via the Tensor
// Data Mover (tensor_load_to_lds + s_wait_tensorcnt), then 13 gather tiles are
// multiplied via WMMA entirely from LDS. Attention + gate epilogue in f32.
// Writes padded f16 sup16 (512, 5, 224) rows (pads pre-zeroed by memset).
// ---------------------------------------------------------------------------
__global__ void encoder_kernel(const int* __restrict__ conn_l,
                               const int* __restrict__ conn_r,
                               const float* __restrict__ sym,
                               const _Float16* __restrict__ gcnW16,  // (112,224)
                               const float* __restrict__ gcn_wb,
                               const float* __restrict__ gcn_b,
                               const float* __restrict__ attnW,
                               const float* __restrict__ attnB,
                               const float* __restrict__ gateW,
                               const float* __restrict__ gateWB,
                               const float* __restrict__ gateB,
                               _Float16* __restrict__ sup16) {
  __shared__ __align__(16) _Float16 Wlds[GCNW_ELEMS];  // 50176 B weight stage
  __shared__ __align__(16) _Float16 Xt[16][KP_GCN];    // gather tile (7168 B)
  __shared__ float outbuf[NBR * 104];                  // 200x100 (stride 104)
  __shared__ float attv[NBR];
  __shared__ float oa[128];
  __shared__ float scal[4];

  int inst = blockIdx.x;
  int side = inst / (FEW * BB);
  int rem  = inst % (FEW * BB);
  int f = rem / BB, b = rem % BB;
  const int* conn = (side == 0 ? conn_l : conn_r) + (long)(f * BB + b) * NBR * 3;
  int tid = threadIdx.x;
  int lane = tid & 31, wave = tid >> 5;
  int half = lane >> 4, l16 = lane & 15;

  // ---- stage padded gcn_W into LDS ----
#if __has_builtin(__builtin_amdgcn_tensor_load_to_lds)
  if (wave == 0) {
    // D# group 0 (ISA 8.3): count=1 | lds_addr | global_addr | type=2
    unsigned long long ga = (unsigned long long)(uintptr_t)gcnW16;
    unsigned int ldsoff = (unsigned int)(uintptr_t)&Wlds[0];  // addr[31:0] = LDS offset
    v4u g0;
    g0.x = 1u;                                        // count=1, user D#
    g0.y = ldsoff;                                    // lds_addr (bytes)
    g0.z = (unsigned int)(ga & 0xFFFFFFFFu);          // global_addr[31:0]
    g0.w = (unsigned int)((ga >> 32) & 0x1FFFFFFu) | (2u << 30);  // [56:32] | type=2
    // D# group 1 (ISA 8.4): data_size=2B, 1 row of 25088 elements
    v8i_ g1;
    g1[0] = 0x00010000;                               // data_size=1 (2 bytes)
    g1[1] = (int)((GCNW_ELEMS & 0xFFFF) << 16);       // tensor_dim0 lo16 @ [31:16]
    g1[2] = (int)(((GCNW_ELEMS >> 16) & 0xFFFF) | (1u << 16)); // dim0 hi | tensor_dim1=1
    g1[3] = (int)((unsigned)GCNW_ELEMS << 16);        // tile_dim0 @ [31:16]
    g1[4] = 1;                                        // tile_dim1=1, tile_dim2=0
    g1[5] = (int)GCNW_ELEMS;                          // tensor_dim0_stride lo32
    g1[6] = 0;
    g1[7] = 0;
    v4i_ gz4 = {0, 0, 0, 0};                          // groups 2/3 unused (2D)
    v8i_ gz8 = {0, 0, 0, 0, 0, 0, 0, 0};              // extra group (clang-23 form)
    __builtin_amdgcn_tensor_load_to_lds(g0, g1, gz4, gz4, gz8, 0);
    __builtin_amdgcn_s_wait_tensorcnt(0);
  }
#else
  for (int i = tid; i < GCNW_ELEMS / 8; i += 256)
    ((v8h*)Wlds)[i] = ((const v8h*)gcnW16)[i];
#endif
  __syncthreads();

  for (int mt = 0; mt < 13; ++mt) {        // 13 tiles cover 200 neighbors
    __syncthreads();
    for (int idx = tid; idx < 16 * KP_GCN; idx += 256) {
      int row = idx / KP_GCN, col = idx % KP_GCN;
      int nb = mt * 16 + row;
      float v = 0.0f;
      if (nb < NBR && col < 2 * DD) {
        int sidx = conn[nb * 3 + (col < DD ? 1 : 2)];
        v = sym[(long)sidx * DD + (col % DD)];
      }
      Xt[row][col] = (_Float16)v;
    }
    __syncthreads();
    if (wave < 7) {                        // 7 N-tiles cover 100 output dims
      int dcol = wave * 16 + l16;          // padded W rows: loads always safe
      const _Float16* pa = &Xt[l16][0] + 8 * half;
      const _Float16* pw = Wlds + (long)dcol * KP_GCN + 8 * half;
      v8f acc = {};
#pragma unroll
      for (int s = 0; s < KP_GCN / 32; ++s) {
        v16h a = frag_f16(pa);
        v16h bm = frag_f16(pw);
        acc = __builtin_amdgcn_wmma_f32_16x16x32_f16(false, a, false, bm,
                                                     (short)0, acc, false, false);
        pa += 32;
        pw += 32;
      }
      if (dcol < DD) {
        float bn = gcn_wb[dcol] + gcn_b[dcol];
#pragma unroll
        for (int r = 0; r < 8; ++r) {
          int nb = mt * 16 + r + 8 * half;
          if (nb < NBR) {
            float v = acc[r] + bn;
            v = (v > 0.0f) ? v : 0.01f * v;     // leaky_relu
            outbuf[nb * 104 + dcol] = v;
          }
        }
      }
    }
  }
  __syncthreads();
  if (tid < NBR) {                          // attention logits
    float l = attnB[0];
    for (int d = 0; d < DD; ++d) l += outbuf[tid * 104 + d] * attnW[d];
    attv[tid] = l;
  }
  __syncthreads();
  if (tid == 0) {
    float mx = -3.4e38f;
    for (int n = 0; n < NBR; ++n) mx = fmaxf(mx, attv[n]);
    scal[0] = mx;
  }
  __syncthreads();
  if (tid < NBR) attv[tid] = __expf(attv[tid] - scal[0]);
  __syncthreads();
  if (tid == 0) {
    float s = 0.0f;
    for (int n = 0; n < NBR; ++n) s += attv[n];
    scal[1] = s;
  }
  __syncthreads();
  if (tid < DD) {                           // weighted sum (softmax folded)
    float s = 0.0f;
    for (int n = 0; n < NBR; ++n) s += attv[n] * outbuf[n * 104 + tid];
    oa[tid] = s / scal[1];
  }
  __syncthreads();
  if (tid == 0) {
    float s = gateWB[0] + gateB[0];
    for (int d = 0; d < DD; ++d) s += oa[d] * gateW[d];
    scal[2] = 1.0f / (1.0f + __expf(-s));
  }
  __syncthreads();
  if (tid < DD) {
    float g = scal[2];
    int entself = conn[0];                  // conn[:,0,0]
    float se = sym[(long)entself * DD + tid];
    float res = oa[tid] * g + se * (1.0f - g);
    sup16[((long)b * FEW + f) * KP_IH0 + side * DD + tid] = (_Float16)res;
  }
}

// ---------------------------------------------------------------------------
// LSTM cell pointwise update (both directions). gpre = (2,512,1800) f32.
// Writes next-h as padded f16 (for the recurrent GEMM) and the sequence
// output: layer0 -> padded f16 y0 (stride 928), layer1 -> f32 y1 (stride 900).
// ---------------------------------------------------------------------------
__global__ void lstm_pointwise(const float* __restrict__ gpre,
                               _Float16* __restrict__ hbuf16,
                               float* __restrict__ cbuf,
                               _Float16* __restrict__ y16,
                               float* __restrict__ y32,
                               float* __restrict__ finals,
                               int layer, int s) {
  int idx = blockIdx.x * blockDim.x + threadIdx.x;
  if (idx >= 2 * BB * HHID) return;
  int dir = idx / (BB * HHID);
  int r = idx % (BB * HHID);
  int b = r / HHID, u = r % HHID;
  const float* g = gpre + ((long)dir * BB + b) * G4;
  float gi = g[u], gf = g[HHID + u], gg = g[2 * HHID + u], go = g[3 * HHID + u];
  float c = cbuf[idx];
  float si = 1.0f / (1.0f + __expf(-gi));
  float sf = 1.0f / (1.0f + __expf(-gf));
  float so = 1.0f / (1.0f + __expf(-go));
  c = sf * c + si * tanhf(gg);
  float h = so * tanhf(c);
  cbuf[idx] = c;
  hbuf16[((long)dir * BB + b) * KP_HH + u] = (_Float16)h;
  int tout = (dir == 0) ? s : (SEQ - 1 - s);
  if (layer == 0)
    y16[((long)tout * BB + b) * KP_IH1 + dir * HHID + u] = (_Float16)h;
  else
    y32[((long)tout * BB + b) * (2 * HHID) + dir * HHID + u] = h;
  if (s == SEQ - 1)
    finals[((long)(2 * layer + dir) * BB + b) * HHID + u] = h;
}

// ---------------------------------------------------------------------------
// Attention over time + context build. Replicates the reference's flat
// reshape of stacked final states: hidden[b][h][l] = finals_flat[b*1800+h*2+l].
// Writes padded f16 context (512 x 1824), pads pre-zeroed.
// ---------------------------------------------------------------------------
__global__ void attn_kernel(const float* __restrict__ y1,
                            const float* __restrict__ finals,
                            _Float16* __restrict__ context16) {
  int b = blockIdx.x;
  int tid = threadIdx.x;                    // 128 threads
  __shared__ float part[128];
  __shared__ float lg[10];
  __shared__ float at[10];
  for (int p = 0; p < 10; ++p) {
    int s = p >> 1, l = p & 1;
    float acc = 0.0f;
    for (int h = tid; h < 2 * HHID; h += 128)
      acc += y1[((long)s * BB + b) * (2 * HHID) + h] *
             finals[(long)b * 1800 + h * 2 + l];
    part[tid] = acc;
    __syncthreads();
    for (int o = 64; o > 0; o >>= 1) {
      if (tid < o) part[tid] += part[tid + o];
      __syncthreads();
    }
    if (tid == 0) lg[p] = part[0];
    __syncthreads();
  }
  if (tid < 2) {
    int l = tid;
    float mx = -3.4e38f;
    for (int s = 0; s < SEQ; ++s) mx = fmaxf(mx, lg[s * 2 + l]);
    float e[SEQ], sum = 0.0f;
    for (int s = 0; s < SEQ; ++s) { e[s] = __expf(lg[s * 2 + l] - mx); sum += e[s]; }
    for (int s = 0; s < SEQ; ++s) at[s * 2 + l] = e[s] / sum;
  }
  __syncthreads();
  for (int h = tid; h < 2 * HHID; h += 128) {
    float c0 = 0.0f, c1 = 0.0f;
    for (int s = 0; s < SEQ; ++s) {
      float yv = y1[((long)s * BB + b) * (2 * HHID) + h];
      c0 += yv * at[s * 2];
      c1 += yv * at[s * 2 + 1];
    }
    context16[(long)b * KP_OUT + h * 2]     = (_Float16)c0;
    context16[(long)b * KP_OUT + h * 2 + 1] = (_Float16)c1;
  }
}

// ---------------------------------------------------------------------------
// Final TransD-style scoring: block per b, wave per candidate j.
// ---------------------------------------------------------------------------
__global__ void score_kernel(const float* __restrict__ hin,
                             const float* __restrict__ tin,
                             const float* __restrict__ normk,
                             const int* __restrict__ head,
                             const int* __restrict__ tail,
                             const int* __restrict__ posn,
                             const float* __restrict__ eemb,
                             const float* __restrict__ rel,
                             float* __restrict__ out) {
  int b = blockIdx.x;
  int tid = threadIdx.x, lane = tid & 31, wave = tid >> 5;
  __shared__ float rr[112], nm[112], rn[1];
  if (tid < DD) { rr[tid] = rel[(long)b * DD + tid]; nm[tid] = normk[(long)b * DD + tid]; }
  __syncthreads();
  if (tid == 0) {
    float s = 0.0f;
    for (int d = 0; d < DD; ++d) s += rr[d] * rr[d];
    rn[0] = fmaxf(sqrtf(s), 1e-12f);
  }
  __syncthreads();
  if (tid < DD) rr[tid] /= rn[0];
  __syncthreads();
  int pn = posn[0];
  for (int j = wave; j < NTOT; j += 8) {
    const float* hp = hin + ((long)b * NTOT + j) * DD;
    const float* tp = tin + ((long)b * NTOT + j) * DD;
    const float* he = eemb + (long)head[b * NTOT + j] * DD;
    const float* te = eemb + (long)tail[b * NTOT + j] * DD;
    float hv[4], tv[4];
    float doth = 0.0f, dott = 0.0f;
#pragma unroll
    for (int q = 0; q < 4; ++q) {
      int d = lane + 32 * q;
      bool v = d < DD;
      float e0 = v ? he[d] : 0.0f, e1 = v ? te[d] : 0.0f;
      hv[q] = v ? hp[d] : 0.0f;
      tv[q] = v ? tp[d] : 0.0f;
      doth += hv[q] * e0;
      dott += tv[q] * e1;
    }
    doth = wave_red_sum(doth);
    dott = wave_red_sum(dott);
    float nsh = 0.0f, nst = 0.0f;
#pragma unroll
    for (int q = 0; q < 4; ++q) {
      int d = lane + 32 * q;
      float nv = (d < DD) ? nm[d] : 0.0f;
      hv[q] += doth * nv;
      tv[q] += dott * nv;
      nsh += hv[q] * hv[q];
      nst += tv[q] * tv[q];
    }
    nsh = wave_red_sum(nsh);
    nst = wave_red_sum(nst);
    float ih = 1.0f / fmaxf(sqrtf(nsh), 1e-12f);
    float it = 1.0f / fmaxf(sqrtf(nst), 1e-12f);
    float n2h = 0.0f, n2t = 0.0f;
#pragma unroll
    for (int q = 0; q < 4; ++q) {
      hv[q] *= ih; tv[q] *= it;
      n2h += hv[q] * hv[q]; n2t += tv[q] * tv[q];
    }
    n2h = wave_red_sum(n2h);                 // second (idempotent) l2norm
    n2t = wave_red_sum(n2t);
    float ih2 = 1.0f / fmaxf(sqrtf(n2h), 1e-12f);
    float it2 = 1.0f / fmaxf(sqrtf(n2t), 1e-12f);
    float ss = 0.0f;
#pragma unroll
    for (int q = 0; q < 4; ++q) {
      int d = lane + 32 * q;
      float rrv = (d < DD) ? rr[d] : 0.0f;
      float diff = hv[q] * ih2 + rrv - tv[q] * it2;
      ss += diff * diff;
    }
    ss = wave_red_sum(ss);
    if (lane == 0) {
      float sc = -sqrtf(ss);
      if (j < pn) out[(long)b * pn + j] = sc;
      else        out[(long)BB * pn + (long)b * (NTOT - pn) + (j - pn)] = sc;
    }
  }
}

// ---------------------------------------------------------------------------
extern "C" void kernel_launch(void* const* d_in, const int* in_sizes, int n_in,
                              void* d_out, int out_size, void* d_ws, size_t ws_size,
                              hipStream_t stream) {
  const int*   conn_l   = (const int*)d_in[0];
  const int*   conn_r   = (const int*)d_in[1];
  const float* h_in     = (const float*)d_in[2];
  const float* t_in     = (const float*)d_in[3];
  const float* norm_in  = (const float*)d_in[4];
  const int*   head     = (const int*)d_in[5];
  const int*   tail     = (const int*)d_in[6];
  const int*   pos_num  = (const int*)d_in[7];
  const float* sym      = (const float*)d_in[8];
  const float* gcnW     = (const float*)d_in[9];
  const float* gcn_wb   = (const float*)d_in[10];
  const float* gcn_b    = (const float*)d_in[11];
  const float* attnW    = (const float*)d_in[12];
  const float* attnB    = (const float*)d_in[13];
  const float* gateW    = (const float*)d_in[14];
  const float* gateWB   = (const float*)d_in[15];
  const float* gateB    = (const float*)d_in[16];
  const float* l0_Wih   = (const float*)d_in[17];
  const float* l0_Whh   = (const float*)d_in[18];
  const float* l0_bih   = (const float*)d_in[19];
  const float* l0_bhh   = (const float*)d_in[20];
  const float* l1_Wih   = (const float*)d_in[21];
  const float* l1_Whh   = (const float*)d_in[22];
  const float* l1_bih   = (const float*)d_in[23];
  const float* l1_bhh   = (const float*)d_in[24];
  const float* out_W    = (const float*)d_in[25];
  const float* out_b    = (const float*)d_in[26];
  const float* eemb     = (const float*)d_in[27];
  float* outp = (float*)d_out;

  // ---- workspace layout ----
  char* wsb = (char*)d_ws;
  size_t off = 0;
  auto carve = [&](size_t bytes) {
    void* p = wsb + off;
    off += (bytes + 255) & ~(size_t)255;
    return p;
  };
  // f32 buffers
  float* y1      = (float*)carve(5 * 512 * 900 * 4);     // (5,512,900)
  float* gpre    = (float*)carve(2 * 512 * 1800 * 4);    // (2,512,1800)
  float* cbuf    = (float*)carve(2 * 512 * 450 * 4);
  float* finals  = (float*)carve(4 * 512 * 450 * 4);
  float* relbuf  = (float*)carve(512 * 100 * 4);
  // padded f16 buffers
  _Float16* sup16     = (_Float16*)carve((size_t)512 * FEW * KP_IH0 * 2);
  _Float16* y0_16     = (_Float16*)carve((size_t)5 * 512 * KP_IH1 * 2);
  _Float16* hbuf16    = (_Float16*)carve((size_t)2 * 512 * KP_HH * 2);
  _Float16* context16 = (_Float16*)carve((size_t)512 * KP_OUT * 2);
  _Float16* gcnW16    = (_Float16*)carve((size_t)NP_SMALL * KP_GCN * 2);
  _Float16* l0Wih16   = (_Float16*)carve((size_t)2 * NP_W * KP_IH0 * 2);
  _Float16* l0Whh16   = (_Float16*)carve((size_t)2 * NP_W * KP_HH * 2);
  _Float16* l1Wih16   = (_Float16*)carve((size_t)2 * NP_W * KP_IH1 * 2);
  _Float16* l1Whh16   = (_Float16*)carve((size_t)2 * NP_W * KP_HH * 2);
  _Float16* outW16    = (_Float16*)carve((size_t)NP_SMALL * KP_OUT * 2);

  auto cvtLaunch = [&](const float* src, int rows, int K, _Float16* dst,
                       int rowsP, int ldk) {
    int n = rowsP * ldk;
    cvt_pad_f16<<<(n + 255) / 256, 256, 0, stream>>>(src, rows, K, dst, rowsP, ldk);
  };
  // ---- 0) weight conversion to padded f16 (pads written as zero) ----
  cvtLaunch(gcnW, 100, 200, gcnW16, NP_SMALL, KP_GCN);
  for (int d = 0; d < 2; ++d) {
    cvtLaunch(l0_Wih + (long)d * 1800 * 200, 1800, 200, l0Wih16 + (long)d * NP_W * KP_IH0, NP_W, KP_IH0);
    cvtLaunch(l0_Whh + (long)d * 1800 * 450, 1800, 450, l0Whh16 + (long)d * NP_W * KP_HH, NP_W, KP_HH);
    cvtLaunch(l1_Wih + (long)d * 1800 * 900, 1800, 900, l1Wih16 + (long)d * NP_W * KP_IH1, NP_W, KP_IH1);
    cvtLaunch(l1_Whh + (long)d * 1800 * 450, 1800, 450, l1Whh16 + (long)d * NP_W * KP_HH, NP_W, KP_HH);
  }
  cvtLaunch(out_W, 100, 1800, outW16, NP_SMALL, KP_OUT);

  // zero dynamic padded f16 activations (pads must be 0 for the GEMMs)
  (void)hipMemsetAsync(sup16, 0, (size_t)512 * FEW * KP_IH0 * 2, stream);
  (void)hipMemsetAsync(y0_16, 0, (size_t)5 * 512 * KP_IH1 * 2, stream);
  (void)hipMemsetAsync(context16, 0, (size_t)512 * KP_OUT * 2, stream);

  // ---- 1) neighbor encoder -> sup16 (B, FEW, 224) f16 ----
  encoder_kernel<<<2 * FEW * BB, 256, 0, stream>>>(
      conn_l, conn_r, sym, gcnW16, gcn_wb, gcn_b,
      attnW, attnB, gateW, gateWB, gateB, sup16);

  // ---- 2) two BiLSTM layers ----
  const int gemmBlocksBig = ((32 * ((1800 + 15) / 16)) + 7) / 8;  // 452
  for (int L = 0; L < 2; ++L) {
    const _Float16* Wih = L ? l1Wih16 : l0Wih16;
    const _Float16* Whh = L ? l1Whh16 : l0Whh16;
    const float* bih = L ? l1_bih : l0_bih;
    const float* bhh = L ? l1_bhh : l0_bhh;
    int KpX  = L ? KP_IH1 : KP_IH0;
    int ldaX = L ? KP_IH1 : LDA_SUP;        // layer0: sup16 rows are (b), stride 1120
    (void)hipMemsetAsync(hbuf16, 0, (size_t)2 * 512 * KP_HH * 2, stream);  // h0=0 + pads
    (void)hipMemsetAsync(cbuf, 0, (size_t)2 * 512 * 450 * 4, stream);      // c0=0
    for (int s = 0; s < SEQ; ++s) {
      const _Float16* A0 = L ? (y0_16 + (long)s * 512 * KP_IH1)
                             : (sup16 + (long)s * KP_IH0);
      const _Float16* A1 = L ? (y0_16 + (long)(4 - s) * 512 * KP_IH1)
                             : (sup16 + (long)(4 - s) * KP_IH0);
      // input projections (+ fused biases)
      gemm_f16wmma<<<gemmBlocksBig, 256, 0, stream>>>(
          A0, ldaX, Wih, KpX, bih, bhh, nullptr, 0,
          gpre, 1800, 512, 1800, KpX);
      gemm_f16wmma<<<gemmBlocksBig, 256, 0, stream>>>(
          A1, ldaX, Wih + (long)NP_W * KpX, KpX, bih + 1800, bhh + 1800, nullptr, 0,
          gpre + 512 * 1800, 1800, 512, 1800, KpX);
      // recurrent projections accumulate into gpre
      gemm_f16wmma<<<gemmBlocksBig, 256, 0, stream>>>(
          hbuf16, KP_HH, Whh, KP_HH, nullptr, nullptr, gpre, 1800,
          gpre, 1800, 512, 1800, KP_HH);
      gemm_f16wmma<<<gemmBlocksBig, 256, 0, stream>>>(
          hbuf16 + (long)512 * KP_HH, KP_HH, Whh + (long)NP_W * KP_HH, KP_HH,
          nullptr, nullptr, gpre + 512 * 1800, 1800,
          gpre + 512 * 1800, 1800, 512, 1800, KP_HH);
      lstm_pointwise<<<(2 * 512 * 450 + 255) / 256, 256, 0, stream>>>(
          gpre, hbuf16, cbuf, y0_16, y1, finals, L, s);
    }
  }

  // ---- 3) attention over time -> context16 (512,1824) f16 ----
  attn_kernel<<<512, 128, 0, stream>>>(y1, finals, context16);

  // ---- 4) rel = context @ out_W.T + out_b -> (512,100) f32 ----
  const int gemmBlocksRel = ((32 * ((100 + 15) / 16)) + 7) / 8;   // 28
  gemm_f16wmma<<<gemmBlocksRel, 256, 0, stream>>>(
      context16, KP_OUT, outW16, KP_OUT, out_b, nullptr, nullptr, 0,
      relbuf, 100, 512, 100, KP_OUT);

  // ---- 5) scoring -> d_out (split at pos_num) ----
  score_kernel<<<512, 256, 0, stream>>>(
      h_in, t_in, norm_in, head, tail, pos_num, eemb, relbuf, outp);
}
